// RETAINLayer_16088947491368
// MI455X (gfx1250) — matrix operand order repaired and checked by reading, not compiled, not run on hardware
//
#include <hip/hip_runtime.h>
#include <hip/hip_bf16.h>

#define Bsz 256
#define Tsz 512
#define Dsz 128
#define D3  384
#define RXS 132   // padded f32 LDS row stride
#define PKS 68    // padded packed-bf16 (u32) LDS row stride (64 words + pad)

typedef float v2f  __attribute__((ext_vector_type(2)));
typedef float v8f  __attribute__((ext_vector_type(8)));
typedef __bf16 v16bf __attribute__((ext_vector_type(16)));

union Frag { v16bf v; unsigned w[8]; };

__device__ __forceinline__ float sigmoidf_(float v) {
    return 1.0f / (1.0f + __expf(-v));
}

// round-to-nearest-even f32 -> bf16, packed pair into one dword
__device__ __forceinline__ unsigned pack_bf16(float a, float b) {
    unsigned ua = __float_as_uint(a);
    unsigned ub = __float_as_uint(b);
    ua += 0x7FFFu + ((ua >> 16) & 1u);
    ub += 0x7FFFu + ((ub >> 16) & 1u);
    return (ua >> 16) | (ub & 0xFFFF0000u);
}

// ---------------------------------------------------------------------------
// Fused (reverse-gather + input-projection + recurrent) GRU kernel.
// One block = 16 batch rows, 24 waves. Wave w owns gate columns [16w,16w+16).
// Matmuls: V_WMMA_F32_16X16X32_BF16, 4 K-chunks (K=128). Weight fragments
// (bf16-packed) live in registers; A-tiles (rx, h) bf16-packed in LDS with a
// one-chunk software pipeline; hidden state kept fp32 in LDS.
// ---------------------------------------------------------------------------
__global__ __launch_bounds__(768, 1)
void retain_gru(const float* __restrict__ x, const int* __restrict__ lens,
                const float* __restrict__ W, const float* __restrict__ U,
                const float* __restrict__ bin, const float* __restrict__ brec,
                float* __restrict__ hs)
{
    __shared__ unsigned s_rxp[16 * PKS];   // bf16-packed reversed-input tile
    __shared__ unsigned s_hp [16 * PKS];   // bf16-packed hidden tile (WMMA A)
    __shared__ float    s_hf [16 * RXS];   // fp32 hidden state
    __shared__ float    s_xzr[16 * 256];   // xz+rz | xr+rr
    __shared__ float    s_xh [16 * 128];
    __shared__ float    s_rh [16 * 128];
    __shared__ int      s_len[16];

    const int tid  = threadIdx.x;
    const int w    = tid >> 5;          // wave 0..23 -> column tile
    const int lane = tid & 31;
    const int nl   = lane & 15;
    const int half = lane >> 4;
    const int b0   = blockIdx.x * 16;
    const int c    = w * 16 + nl;       // gate column 0..383
    const int m    = nl;                // A-row for this lane

    if (tid < 16) s_len[tid] = lens[b0 + tid];
    for (int i = tid; i < 16 * RXS; i += 768) s_hf[i] = 0.f;
    for (int i = tid; i < 16 * PKS; i += 768) s_hp[i] = 0u;

    // Preload bf16 B-operand fragments. 16x16x32 B layout: lane L -> col L%16;
    // VGPR v<4 holds K = 8*half + 2v, +1; v>=4 holds K = 16 + 8*half + 2(v-4), +1.
    Frag wf[4], uf[4];
#pragma unroll
    for (int kc = 0; kc < 4; ++kc) {
#pragma unroll
        for (int v = 0; v < 8; ++v) {
            const int k = kc * 32 + (v < 4 ? half * 8 + 2 * v : 16 + half * 8 + 2 * (v - 4));
            wf[kc].w[v] = pack_bf16(W[(size_t)k * D3 + c], W[(size_t)(k + 1) * D3 + c]);
            uf[kc].w[v] = pack_bf16(U[(size_t)k * D3 + c], U[(size_t)(k + 1) * D3 + c]);
        }
    }
    const float bin_c  = bin[c];
    const float brec_c = brec[c];

    // Gate-staging base offsets (wave-uniform branch hoisted out of p-loop)
    const bool wlow   = (w < 16);
    const int  o_xzr  = half * 8 * 256 + c;          // valid when wlow
    const int  o_h    = half * 8 * 128 + (c - 256);  // valid when !wlow

    // Fixed per-thread slots over the 16x64 packed-word tile (1024 words):
    // slot0 = tid, slot1 = tid+768 (threads < 256). Pointers march with t.
    const int r0 = tid >> 6, q0 = tid & 63;
    const int L0 = lens[b0 + r0];
    const float* px0 = x + ((size_t)(b0 + r0) * Tsz + (size_t)(L0 - 1)) * Dsz + 2 * q0;
    const int so0 = r0 * PKS + q0;
    float* ph0 = hs + (size_t)(b0 + r0) * Dsz + 2 * q0;

    const bool has1 = tid < 256;
    int r1 = 0, q1 = 0, L1 = 1, so1 = 0;
    const float* px1 = x;
    float* ph1 = hs;
    if (has1) {
        const int i = tid + 768;
        r1 = i >> 6; q1 = i & 63;
        L1 = lens[b0 + r1];
        px1 = x + ((size_t)(b0 + r1) * Tsz + (size_t)(L1 - 1)) * Dsz + 2 * q1;
        so1 = r1 * PKS + q1;
        ph1 = hs + (size_t)(b0 + r1) * Dsz + 2 * q1;
    }
    __syncthreads();

    for (int t = 0; t < Tsz; ++t) {
        // 1) gather reversed, masked input; pack to bf16 pairs
        {
            float a = 0.f, b = 0.f;
            if (t < L0) { a = px0[0]; b = px0[1]; }
            s_rxp[so0] = pack_bf16(a, b);
            px0 -= Dsz;
            if (has1) {
                float a1 = 0.f, b1 = 0.f;
                if (t < L1) { a1 = px1[0]; b1 = px1[1]; }
                s_rxp[so1] = pack_bf16(a1, b1);
                px1 -= Dsz;
            }
        }
        __syncthreads();   // sync1: rx ready; prev gate writes to s_hp visible

        // 2) xp = rx@W, rp = h@U  (bf16 WMMA, fp32 accumulate, 1-chunk pipeline)
        v8f accX = {0.f, 0.f, 0.f, 0.f, 0.f, 0.f, 0.f, 0.f};
        v8f accR = {0.f, 0.f, 0.f, 0.f, 0.f, 0.f, 0.f, 0.f};
        {
            const int base0 = m * PKS + half * 4;
            Frag ax, ah, axn, ahn;
            *(uint4*)&ax.w[0] = *(const uint4*)&s_rxp[base0];
            *(uint4*)&ax.w[4] = *(const uint4*)&s_rxp[base0 + 8];
            *(uint4*)&ah.w[0] = *(const uint4*)&s_hp[base0];
            *(uint4*)&ah.w[4] = *(const uint4*)&s_hp[base0 + 8];
#pragma unroll
            for (int kc = 0; kc < 4; ++kc) {
                if (kc < 3) {
                    const int nb = base0 + (kc + 1) * 16;
                    *(uint4*)&axn.w[0] = *(const uint4*)&s_rxp[nb];
                    *(uint4*)&axn.w[4] = *(const uint4*)&s_rxp[nb + 8];
                    *(uint4*)&ahn.w[0] = *(const uint4*)&s_hp[nb];
                    *(uint4*)&ahn.w[4] = *(const uint4*)&s_hp[nb + 8];
                }
                accX = __builtin_amdgcn_wmma_f32_16x16x32_bf16(
                           false, ax.v, false, wf[kc].v, (short)0, accX, false, false);
                accR = __builtin_amdgcn_wmma_f32_16x16x32_bf16(
                           false, ah.v, false, uf[kc].v, (short)0, accR, false, false);
                ax = axn; ah = ahn;
            }
        }

        // 3) stage gate pre-activations (uniform branch hoisted out of p-loop)
        if (wlow) {
#pragma unroll
            for (int p = 0; p < 8; ++p)
                s_xzr[o_xzr + p * 256] = (accX[p] + bin_c) + (accR[p] + brec_c);
        } else {
#pragma unroll
            for (int p = 0; p < 8; ++p) {
                s_xh[o_h + p * 128] = accX[p] + bin_c;
                s_rh[o_h + p * 128] = accR[p] + brec_c;
            }
        }
        __syncthreads();   // sync2: gate bufs ready; WMMA reads of s_rxp/s_hp done

        // 4) elementwise GRU update (fp32 state), emit hs[t], repack bf16
        {
            const int xb = r0 * 256 + 2 * q0;
            const int hb = r0 * 128 + 2 * q0;
            const int fb = r0 * RXS + 2 * q0;
            const float z0  = sigmoidf_(s_xzr[xb]);
            const float z1  = sigmoidf_(s_xzr[xb + 1]);
            const float rg0 = sigmoidf_(s_xzr[xb + 128]);
            const float rg1 = sigmoidf_(s_xzr[xb + 129]);
            const float hh0 = tanhf(s_xh[hb] + rg0 * s_rh[hb]);
            const float hh1 = tanhf(s_xh[hb + 1] + rg1 * s_rh[hb + 1]);
            const float ho0 = s_hf[fb], ho1 = s_hf[fb + 1];
            float hn0 = z0 * ho0 + (1.f - z0) * hh0;
            float hn1 = z1 * ho1 + (1.f - z1) * hh1;
            if (t >= L0) { hn0 = ho0; hn1 = ho1; }
            s_hf[fb] = hn0; s_hf[fb + 1] = hn1;
            s_hp[r0 * PKS + q0] = pack_bf16(hn0, hn1);
            ph0[0] = hn0; ph0[1] = hn1;
            ph0 += (size_t)Bsz * Dsz;
            if (has1) {
                const int xb1 = r1 * 256 + 2 * q1;
                const int hb1 = r1 * 128 + 2 * q1;
                const int fb1 = r1 * RXS + 2 * q1;
                const float z0b  = sigmoidf_(s_xzr[xb1]);
                const float z1b  = sigmoidf_(s_xzr[xb1 + 1]);
                const float rg0b = sigmoidf_(s_xzr[xb1 + 128]);
                const float rg1b = sigmoidf_(s_xzr[xb1 + 129]);
                const float hh0b = tanhf(s_xh[hb1] + rg0b * s_rh[hb1]);
                const float hh1b = tanhf(s_xh[hb1 + 1] + rg1b * s_rh[hb1 + 1]);
                const float ho0b = s_hf[fb1], ho1b = s_hf[fb1 + 1];
                float hn0b = z0b * ho0b + (1.f - z0b) * hh0b;
                float hn1b = z1b * ho1b + (1.f - z1b) * hh1b;
                if (t >= L1) { hn0b = ho0b; hn1b = ho1b; }
                s_hf[fb1] = hn0b; s_hf[fb1 + 1] = hn1b;
                s_hp[r1 * PKS + q1] = pack_bf16(hn0b, hn1b);
                ph1[0] = hn0b; ph1[1] = hn1b;
                ph1 += (size_t)Bsz * Dsz;
            }
        }
        // next-iter sync1 orders s_hp/s_hf reuse
    }
}

// ---------------------------------------------------------------------------
// alpha = softmax_T(g @ W_alpha + b_alpha): one block per sample.
// ---------------------------------------------------------------------------
__global__ __launch_bounds__(512, 1)
void retain_alpha(const float* __restrict__ g, const float* __restrict__ Wa,
                  const float* __restrict__ ba, float* __restrict__ alpha)
{
    __shared__ float red[512];
    const int b = blockIdx.x, t = threadIdx.x;
    const float* gp = g + ((size_t)t * Bsz + b) * Dsz;
    float e = ba[0];
#pragma unroll 4
    for (int d = 0; d < Dsz; ++d) e += gp[d] * Wa[d];
    red[t] = e;
    __syncthreads();
    for (int s = 256; s > 0; s >>= 1) {
        if (t < s) red[t] = fmaxf(red[t], red[t + s]);
        __syncthreads();
    }
    const float mx = red[0];
    __syncthreads();
    const float ex = __expf(e - mx);
    red[t] = ex;
    __syncthreads();
    for (int s = 256; s > 0; s >>= 1) {
        if (t < s) red[t] += red[t + s];
        __syncthreads();
    }
    alpha[(size_t)b * Tsz + t] = ex / red[0];
}

// ---------------------------------------------------------------------------
// out[b,:] += sum_t alpha[b,t] * tanh(h[b,t,:]@W_beta + b_beta) * x[b,t,:]
// fp32 WMMA (one-shot, accuracy-sensitive, no register pressure issue).
// ---------------------------------------------------------------------------
__global__ __launch_bounds__(256, 1)
void retain_attn_out(const float* __restrict__ x, const float* __restrict__ h,
                     const float* __restrict__ alpha,
                     const float* __restrict__ Wb, const float* __restrict__ bb,
                     float* __restrict__ out)
{
    __shared__ float s_a[16 * RXS];
    const int tid  = threadIdx.x;
    const int w    = tid >> 5;          // 0..7
    const int lane = tid & 31;
    const int nl   = lane & 15;
    const int half = lane >> 4;
    const int b    = blockIdx.x;
    const int t0   = blockIdx.y * 16;
    const int c    = w * 16 + nl;       // output column 0..127

    for (int i = tid; i < 16 * 128; i += 256) {
        const int r = i >> 7, j = i & 127;
        s_a[r * RXS + j] = h[((size_t)(t0 + r) * Bsz + b) * Dsz + j];
    }
    __syncthreads();

    v8f acc = {0.f, 0.f, 0.f, 0.f, 0.f, 0.f, 0.f, 0.f};
#pragma unroll
    for (int kc = 0; kc < 32; ++kc) {
        const int k0 = kc * 4 + half * 2;
        v2f a, bf;
        a.x  = s_a[nl * RXS + k0];
        a.y  = s_a[nl * RXS + k0 + 1];
        bf.x = Wb[(size_t)k0 * Dsz + c];
        bf.y = Wb[(size_t)(k0 + 1) * Dsz + c];
        acc = __builtin_amdgcn_wmma_f32_16x16x4_f32(
                  false, a, false, bf, (short)0, acc, false, false);
    }

    const float bbias = bb[c];
    float s = 0.f;
#pragma unroll
    for (int p = 0; p < 8; ++p) {
        const int row = p + half * 8;
        const int t   = t0 + row;
        const float bv = tanhf(acc[p] + bbias);
        const float al = alpha[(size_t)b * Tsz + t];
        const float xv = x[((size_t)b * Tsz + t) * Dsz + c];
        s += al * bv * xv;
    }
    s += __shfl_xor(s, 16, 32);         // combine row-halves
    if (half == 0) atomicAdd(&out[(size_t)b * Dsz + c], s);
}

__global__ void retain_zero(float* __restrict__ p, int n)
{
    const int i = blockIdx.x * blockDim.x + threadIdx.x;
    if (i < n) p[i] = 0.f;
}

// ---------------------------------------------------------------------------
extern "C" void kernel_launch(void* const* d_in, const int* in_sizes, int n_in,
                              void* d_out, int out_size, void* d_ws, size_t ws_size,
                              hipStream_t stream)
{
    const float* x      = (const float*)d_in[0];
    const int*   lens   = (const int*)  d_in[1];
    const float* Wa     = (const float*)d_in[2];
    const float* Ua     = (const float*)d_in[3];
    const float* ba_in  = (const float*)d_in[4];
    const float* ba_rec = (const float*)d_in[5];
    const float* Wb     = (const float*)d_in[6];
    const float* Ub     = (const float*)d_in[7];
    const float* bb_in  = (const float*)d_in[8];
    const float* bb_rec = (const float*)d_in[9];
    const float* W_al   = (const float*)d_in[10];
    const float* b_al   = (const float*)d_in[11];
    const float* W_be   = (const float*)d_in[12];
    const float* b_be   = (const float*)d_in[13];
    float* out = (float*)d_out;

    float* g     = (float*)d_ws;                    // [T,B,D]
    float* h     = g + (size_t)Tsz * Bsz * Dsz;     // [T,B,D]
    float* alpha = h + (size_t)Tsz * Bsz * Dsz;     // [B,T]

    retain_zero<<<dim3((Bsz * Dsz + 255) / 256), dim3(256), 0, stream>>>(out, Bsz * Dsz);

    retain_gru<<<dim3(Bsz / 16), dim3(768), 0, stream>>>(
        x, lens, Wa, Ua, ba_in, ba_rec, g);
    retain_gru<<<dim3(Bsz / 16), dim3(768), 0, stream>>>(
        x, lens, Wb, Ub, bb_in, bb_rec, h);

    retain_alpha<<<dim3(Bsz), dim3(Tsz), 0, stream>>>(g, W_al, b_al, alpha);

    retain_attn_out<<<dim3(Bsz, Tsz / 16), dim3(256), 0, stream>>>(
        x, h, alpha, W_be, b_be, out);
}
